// GLANTConv_63943473102986
// MI455X (gfx1250) — matrix operand order repaired and checked by reading, not compiled, not run on hardware
//
#include <hip/hip_runtime.h>
#include <hip/hip_bf16.h>
#include <math.h>

typedef __attribute__((ext_vector_type(16))) _Float16 v16h;
typedef __attribute__((ext_vector_type(8)))  _Float16 v8h;
typedef __attribute__((ext_vector_type(8)))  float    v8f;

#define IN_CH 256
#define HC    256      // H*C
#define HEADS 4
#define CPH   64
#define NEG_SLOPE 0.2f
#define SK 264         // padded K stride (in halves) for LDS W tile

// ---------------------------------------------------------------------------
// Fused dual GEMM: YL = X @ WL, YR = X @ WR  (f16 WMMA, f32 accumulate)
// A fragments (x rows) are loaded/converted ONCE and reused for both weights.
// Block: 256 threads = 8 waves; each wave -> 16 rows x 64 cols x 2 outputs.
// grid = (ceil(N/128), 4 column chunks). LDS: one 64-col weight tile (33.8KB),
// restaged between the WL and WR phases behind a barrier.
// ---------------------------------------------------------------------------
__global__ __launch_bounds__(256)
void gemm_proj2(const float* __restrict__ X,
                const float* __restrict__ WL, const float* __restrict__ WR,
                float* __restrict__ YL, float* __restrict__ YR, int nRows)
{
    __shared__ _Float16 ldsW[64 * SK];   // 64 cols x 256 K, padded
    const int tid  = threadIdx.x;
    const int lane = tid & 31;
    const int wave = tid >> 5;
    const int n0   = blockIdx.y * 64;
    const int r0   = blockIdx.x * 128 + wave * 16;

    const int rowA  = min(r0 + (lane & 15), nRows - 1);
    const int kgrpA = (lane >> 4) << 4;   // A frag: +0 or +16
    const int kgrpB = (lane >> 4) << 4;   // B frag: +0 or +16
    const float* xrow = X + (size_t)rowA * IN_CH;

    // ---- Load all 8 A fragments once (contiguous k order, matching B):
    // lane<16: row M=lane, K k0+0..15 contiguous; lane>=16: K k0+16..31.
    v16h a[8];
    #pragma unroll
    for (int kt = 0; kt < 8; ++kt) {
        const float* xp = xrow + kt * 32 + kgrpA;
        float buf[16];
        *(float4*)(buf + 0)  = *(const float4*)(xp + 0);
        *(float4*)(buf + 4)  = *(const float4*)(xp + 4);
        *(float4*)(buf + 8)  = *(const float4*)(xp + 8);
        *(float4*)(buf + 12) = *(const float4*)(xp + 12);
        #pragma unroll
        for (int j = 0; j < 16; ++j) a[kt][j] = (_Float16)buf[j];
    }

    const int colBase = n0 + (lane & 15);
    const int mOff    = (lane < 16) ? 0 : 8;

    const float* Wmat[2] = { WL, WR };
    float*       Yout[2] = { YL, YR };

    #pragma unroll
    for (int p = 0; p < 2; ++p) {
        __syncthreads();   // all B reads of previous phase done before restage
        // Stage W[:, n0:n0+64] into LDS transposed (ldsW[n*SK + k]), f32->f16.
        const float* Wp = Wmat[p];
        for (int i = tid; i < 64 * 256; i += 256) {
            int n = i & 63;
            int k = i >> 6;
            ldsW[n * SK + k] = (_Float16)Wp[k * 256 + n0 + n];
        }
        __syncthreads();

        v8f acc[4] = {};
        #pragma unroll
        for (int kt = 0; kt < 8; ++kt) {
            const int k0 = kt * 32;
            // Hoist all 4 B fragments: ds_loads issue as a clause, one wait
            // covers 4 WMMAs. B layout: lane<16 -> col N=lane, K k0+0..15
            // contiguous; lane>=16 -> same cols, K k0+16..31.
            v8h blo[4], bhi[4];
            #pragma unroll
            for (int nt = 0; nt < 4; ++nt) {
                const v8h* bp = (const v8h*)
                    &ldsW[(nt * 16 + (lane & 15)) * SK + k0 + kgrpB];
                blo[nt] = bp[0];
                bhi[nt] = bp[1];
            }
            #pragma unroll
            for (int nt = 0; nt < 4; ++nt) {
                v16h b = __builtin_shufflevector(blo[nt], bhi[nt],
                         0,1,2,3,4,5,6,7,8,9,10,11,12,13,14,15);
                acc[nt] = __builtin_amdgcn_wmma_f32_16x16x32_f16(
                    false, a[kt], false, b, (short)0, acc[nt], false, false);
            }
        }

        // D layout: lane holds col N=lane&15; VGPR r -> row M=r+(lane<16?0:8)
        float* Yp = Yout[p];
        #pragma unroll
        for (int nt = 0; nt < 4; ++nt) {
            #pragma unroll
            for (int r = 0; r < 8; ++r) {
                int row = r0 + mOff + r;
                if (row < nRows)
                    Yp[(size_t)row * HC + colBase + nt * 16] = acc[nt][r];
            }
        }
    }
}

// ---------------------------------------------------------------------------
// Order-preserving uint encoding of f32 (for atomicMax-based segment max)
// ---------------------------------------------------------------------------
__device__ __forceinline__ unsigned fenc(float f) {
    unsigned u = __float_as_uint(f);
    return (u >> 31) ? ~u : (u | 0x80000000u);
}
__device__ __forceinline__ float fdec(unsigned e) {
    unsigned u = (e & 0x80000000u) ? (e & 0x7FFFFFFFu) : ~e;
    return __uint_as_float(u);
}

// ---------------------------------------------------------------------------
// Pass 1: edge score e[e,h] = att[h,:] . leaky_relu(XL[src]+XR[dst]); seg-max
// One wave per edge; lane covers 8 contiguous channels (one head per 8 lanes)
// ---------------------------------------------------------------------------
__global__ __launch_bounds__(256)
void gat_score(const float* __restrict__ XL, const float* __restrict__ XR,
               const float* __restrict__ att, const int* __restrict__ ei,
               int nE, int nList, float* __restrict__ esc,
               unsigned* __restrict__ nmax)
{
    int eid = blockIdx.x * 8 + (threadIdx.x >> 5);
    if (eid >= nE) return;
    int lane = threadIdx.x & 31;
    int src, dst;
    if (eid < nList) { src = ei[eid]; dst = ei[nList + eid]; }
    else             { src = dst = eid - nList; }   // self loop (hop 0)
    int f0 = lane * 8;
    const float* pl = XL + (size_t)src * HC + f0;
    const float* pr = XR + (size_t)dst * HC + f0;
    float xl[8], xr[8], aw[8];
    *(float4*)(xl)     = *(const float4*)(pl);
    *(float4*)(xl + 4) = *(const float4*)(pl + 4);
    *(float4*)(xr)     = *(const float4*)(pr);
    *(float4*)(xr + 4) = *(const float4*)(pr + 4);
    *(float4*)(aw)     = *(const float4*)(att + f0);
    *(float4*)(aw + 4) = *(const float4*)(att + f0 + 4);
    float s = 0.f;
    #pragma unroll
    for (int j = 0; j < 8; ++j) {
        float v = xl[j] + xr[j];
        v = (v > 0.f) ? v : NEG_SLOPE * v;
        s += aw[j] * v;
    }
    #pragma unroll
    for (int off = 1; off < 8; off <<= 1) s += __shfl_xor(s, off, 32);
    if ((lane & 7) == 0) {
        int h = lane >> 3;
        esc[(size_t)eid * HEADS + h] = s;
        atomicMax(&nmax[dst * HEADS + h], fenc(s));
    }
}

// ---------------------------------------------------------------------------
// Pass 2: ex = exp(e - max[dst]); segment-sum denominators
// ---------------------------------------------------------------------------
__global__ __launch_bounds__(256)
void gat_expsum(float* __restrict__ esc, const int* __restrict__ ei,
                int nE, int nList, const unsigned* __restrict__ nmax,
                float* __restrict__ denom)
{
    int idx = blockIdx.x * blockDim.x + threadIdx.x;
    if (idx >= nE * HEADS) return;
    int eid = idx >> 2;
    int h   = idx & 3;
    int dst = (eid < nList) ? ei[nList + eid] : (eid - nList);
    float m  = fdec(nmax[dst * HEADS + h]);
    float ex = expf(esc[idx] - m);
    esc[idx] = ex;
    atomicAdd(&denom[dst * HEADS + h], ex);
}

// ---------------------------------------------------------------------------
// Pass 3: out[dst] += alpha * XL[src]  (one wave per edge, f32 atomics)
// ---------------------------------------------------------------------------
__global__ __launch_bounds__(256)
void gat_aggregate(const float* __restrict__ XL, const float* __restrict__ esc,
                   const float* __restrict__ denom, const int* __restrict__ ei,
                   int nE, int nList, float* __restrict__ accum)
{
    int eid = blockIdx.x * 8 + (threadIdx.x >> 5);
    if (eid >= nE) return;
    int lane = threadIdx.x & 31;
    int src, dst;
    if (eid < nList) { src = ei[eid]; dst = ei[nList + eid]; }
    else             { src = dst = eid - nList; }
    int h = lane >> 3;
    float alpha = esc[(size_t)eid * HEADS + h] /
                  (denom[dst * HEADS + h] + 1e-16f);
    int f0 = lane * 8;
    const float* pl = XL + (size_t)src * HC + f0;
    float xl[8];
    *(float4*)(xl)     = *(const float4*)(pl);
    *(float4*)(xl + 4) = *(const float4*)(pl + 4);
    float* pa = accum + (size_t)dst * HC + f0;
    #pragma unroll
    for (int j = 0; j < 8; ++j) atomicAdd(pa + j, xl[j] * alpha);
}

// ---------------------------------------------------------------------------
// Pass 4: head-mean, hop combine: out = mean_h(accum) [* scale, += if add]
// ---------------------------------------------------------------------------
__global__ __launch_bounds__(256)
void gat_finalize(const float* __restrict__ accum, float* __restrict__ out,
                  int nNodes, float scale, int add)
{
    int idx = blockIdx.x * blockDim.x + threadIdx.x;
    if (idx >= nNodes * CPH) return;
    int n = idx >> 6;
    int c = idx & 63;
    const float* p = accum + (size_t)n * HC + c;
    float v = (p[0] + p[64] + p[128] + p[192]) * 0.25f * scale;
    if (add) out[idx] += v; else out[idx] = v;
}

__global__ void zero_f32(float* __restrict__ p, size_t n)
{
    size_t i = (size_t)blockIdx.x * blockDim.x + threadIdx.x;
    size_t stride = (size_t)gridDim.x * blockDim.x;
    for (; i < n; i += stride) p[i] = 0.f;
}

// ---------------------------------------------------------------------------
extern "C" void kernel_launch(void* const* d_in, const int* in_sizes, int n_in,
                              void* d_out, int out_size, void* d_ws, size_t ws_size,
                              hipStream_t stream)
{
    const float* x    = (const float*)d_in[0];
    const float* Wl0  = (const float*)d_in[1];
    const float* Wr0  = (const float*)d_in[2];
    const float* att0 = (const float*)d_in[3];
    const float* Wl1  = (const float*)d_in[4];
    const float* Wr1  = (const float*)d_in[5];
    const float* att1 = (const float*)d_in[6];
    const int*   ei1  = (const int*)d_in[7];
    const int*   ei2  = (const int*)d_in[8];
    const int N = in_sizes[0] / IN_CH;
    const int E = in_sizes[7] / 2;
    float* out = (float*)d_out;

    // workspace layout
    char* ws = (char*)d_ws;
    size_t offXL    = 0;
    size_t offXR    = offXL  + (size_t)N * HC * sizeof(float);
    size_t offEsc   = offXR  + (size_t)N * HC * sizeof(float);
    size_t offNmax  = offEsc + (size_t)(E + N) * HEADS * sizeof(float);
    size_t offDenom = offNmax + (size_t)N * HEADS * sizeof(unsigned);
    float*    XL    = (float*)(ws + offXL);
    float*    XR    = (float*)(ws + offXR);    // reused as accumulator
    float*    esc   = (float*)(ws + offEsc);
    unsigned* nmax  = (unsigned*)(ws + offNmax);
    float*    denom = (float*)(ws + offDenom); // contiguous after nmax

    dim3 gGemm((N + 127) / 128, 4);
    for (int hop = 0; hop < 2; ++hop) {
        const float* Wl  = hop ? Wl1  : Wl0;
        const float* Wr  = hop ? Wr1  : Wr0;
        const float* att = hop ? att1 : att0;
        const int*   ei  = hop ? ei2  : ei1;
        const int    nE  = hop ? E : (E + N);   // hop 0 adds self loops

        gemm_proj2<<<gGemm, 256, 0, stream>>>(x, Wl, Wr, XL, XR, N);
        // zero nmax + denom (contiguous region)
        zero_f32<<<256, 256, 0, stream>>>((float*)nmax, (size_t)N * HEADS * 2);
        gat_score<<<(nE + 7) / 8, 256, 0, stream>>>(XL, XR, att, ei, nE, E,
                                                    esc, nmax);
        gat_expsum<<<(nE * HEADS + 255) / 256, 256, 0, stream>>>(esc, ei, nE, E,
                                                                 nmax, denom);
        // XR no longer needed -> becomes the aggregation accumulator
        zero_f32<<<512, 256, 0, stream>>>(XR, (size_t)N * HC);
        gat_aggregate<<<(nE + 7) / 8, 256, 0, stream>>>(XL, esc, denom, ei,
                                                        nE, E, XR);
        gat_finalize<<<(N * CPH + 255) / 256, 256, 0, stream>>>(
            XR, out, N, hop ? 0.5f : 1.0f, hop);
    }
}
